// SelfAttention4d_52003464020120
// MI455X (gfx1250) — compile-verified, matrix operand order checked by reading, not run on hardware
//
#include <hip/hip_runtime.h>

#define EMBED 512
#define NTOK  4096
#define HEADS 8
#define HDIM  64

typedef __bf16        bf16x16 __attribute__((ext_vector_type(16)));
typedef float         f32x8   __attribute__((ext_vector_type(8)));
typedef unsigned int  u32x8   __attribute__((ext_vector_type(8)));

#if defined(__AMDGCN__) && __has_builtin(__builtin_amdgcn_global_load_async_to_lds_b128)
#define HAVE_ASYNC_LDS 1
typedef int v4i_ __attribute__((vector_size(16)));
typedef __attribute__((address_space(1))) v4i_ v4i_glb;
typedef __attribute__((address_space(3))) v4i_ v4i_lds;
#else
#define HAVE_ASYNC_LDS 0
#endif

// ---- helpers -------------------------------------------------------------

__device__ __forceinline__ unsigned short f2bf(float x) {
  unsigned int u = __builtin_bit_cast(unsigned int, x);
  u += 0x7FFFu + ((u >> 16) & 1u);        // round-to-nearest-even
  return (unsigned short)(u >> 16);
}

__device__ __forceinline__ unsigned int pack2bf(float v0, float v1) {
#if defined(__AMDGCN__) && __has_builtin(__builtin_amdgcn_cvt_pk_bf16_f32)
  typedef __bf16 bf16x2 __attribute__((ext_vector_type(2)));
  bf16x2 p = __builtin_amdgcn_cvt_pk_bf16_f32(v0, v1);
  return __builtin_bit_cast(unsigned int, p);
#else
  return (unsigned int)f2bf(v0) | ((unsigned int)f2bf(v1) << 16);
#endif
}

__device__ __forceinline__ f32x8 zero8() {
  f32x8 v;
#pragma unroll
  for (int i = 0; i < 8; ++i) v[i] = 0.f;
  return v;
}

// CDNA5 16-bit A/B fragment (16x32 / 32x16): lane L holds row/col (L&15),
// K elements [k0+ko, k0+ko+8) and [k0+ko+16, k0+ko+24) with ko = (L>>4)*8.
// With K contiguous in memory this is exactly two b128 loads.
__device__ __forceinline__ bf16x16 load_frag(const unsigned short* row, int k0, int lh) {
  const uint4* p0 = reinterpret_cast<const uint4*>(row + k0 + lh * 8);
  const uint4* p1 = reinterpret_cast<const uint4*>(row + k0 + lh * 8 + 16);
  uint4 a = *p0;
  uint4 b = *p1;
  u32x8 u;
  u[0] = a.x; u[1] = a.y; u[2] = a.z; u[3] = a.w;
  u[4] = b.x; u[5] = b.y; u[6] = b.z; u[7] = b.w;
  return __builtin_bit_cast(bf16x16, u);
}

__device__ __forceinline__ f32x8 wmma_bf16(bf16x16 a, bf16x16 b, f32x8 c) {
  return __builtin_amdgcn_wmma_f32_16x16x32_bf16(false, a, false, b, (short)0, c, false, false);
}

// ---- conversion kernels --------------------------------------------------

__global__ __launch_bounds__(256) void convert_scale_k(const float* __restrict__ src,
                                                       unsigned short* __restrict__ dst,
                                                       int n, float scale) {
  int i = blockIdx.x * 256 + threadIdx.x;
  if (i < n) dst[i] = f2bf(src[i] * scale);
}

// hidden_states (C=512 rows x N=4096 cols, f32) -> Xt[n][c] bf16 (K contiguous)
__global__ __launch_bounds__(256) void transpose_convert_k(const float* __restrict__ src,
                                                           unsigned short* __restrict__ dst) {
  int i = blockIdx.x * 256 + threadIdx.x;   // over 512*4096
  int c = i >> 12;
  int n = i & 4095;
  dst[(size_t)n * EMBED + c] = f2bf(src[i]);
}

// ---- GEMM: Out = W(MxK) * X(NxK)^T + bias --------------------------------
// OUT_MODE 0: bf16, transposed  Out[n][m]   (for Q, K)
// OUT_MODE 1: bf16, normal      Out[m][n]   (for V)
// OUT_MODE 2: f32,  normal      Out[m][n]   (final projection)
template <int OUT_MODE>
__global__ __launch_bounds__(256) void gemm_wmma(const unsigned short* __restrict__ W,
                                                 const unsigned short* __restrict__ X,
                                                 const float* __restrict__ bias,
                                                 float bias_scale,
                                                 void* __restrict__ OutP) {
  const int tid  = threadIdx.x;
  const int w    = tid >> 5;
  const int lane = tid & 31;
  const int li   = lane & 15;
  const int lh   = lane >> 4;
  const int wm   = w >> 2;                       // 0..1
  const int wn   = w & 3;                        // 0..3
  const int mb   = blockIdx.x * 128 + wm * 64;   // wave: 64 x 32 tile
  const int nb   = blockIdx.y * 128 + wn * 32;

  f32x8 acc[4][2];
#pragma unroll
  for (int i = 0; i < 4; ++i)
#pragma unroll
    for (int j = 0; j < 2; ++j) acc[i][j] = zero8();

  for (int k0 = 0; k0 < EMBED; k0 += 32) {
    bf16x16 af[4], bf[2];
#pragma unroll
    for (int i = 0; i < 4; ++i)
      af[i] = load_frag(W + (size_t)(mb + i * 16 + li) * EMBED, k0, lh);
#pragma unroll
    for (int j = 0; j < 2; ++j)
      bf[j] = load_frag(X + (size_t)(nb + j * 16 + li) * EMBED, k0, lh);
#pragma unroll
    for (int i = 0; i < 4; ++i)
#pragma unroll
      for (int j = 0; j < 2; ++j) acc[i][j] = wmma_bf16(af[i], bf[j], acc[i][j]);
  }

#pragma unroll
  for (int i = 0; i < 4; ++i)
#pragma unroll
    for (int j = 0; j < 2; ++j) {
      const int n  = nb + j * 16 + li;
      const int m0 = mb + i * 16 + lh * 8;
      if constexpr (OUT_MODE == 0) {
        unsigned short* O = (unsigned short*)OutP;
#pragma unroll
        for (int r = 0; r < 8; r += 2) {
          float v0 = acc[i][j][r]     + bias[m0 + r]     * bias_scale;
          float v1 = acc[i][j][r + 1] + bias[m0 + r + 1] * bias_scale;
          *reinterpret_cast<unsigned int*>(O + (size_t)n * EMBED + m0 + r) = pack2bf(v0, v1);
        }
      } else if constexpr (OUT_MODE == 1) {
        unsigned short* O = (unsigned short*)OutP;
#pragma unroll
        for (int r = 0; r < 8; ++r)
          O[(size_t)(m0 + r) * NTOK + n] = f2bf(acc[i][j][r] + bias[m0 + r] * bias_scale);
      } else {
        float* O = (float*)OutP;
#pragma unroll
        for (int r = 0; r < 8; ++r)
          O[(size_t)(m0 + r) * NTOK + n] = acc[i][j][r] + bias[m0 + r] * bias_scale;
      }
    }
}

// ---- flash attention -----------------------------------------------------
// Grid: (NTOK/128, HEADS). 8 waves; wave owns 16 queries x full d=64.
// Qt/Kt: [n][c] bf16 (d contiguous). Vc: [c][n] bf16 (keys contiguous).
__global__ __launch_bounds__(256) void flash_attn(const unsigned short* __restrict__ Qt,
                                                  const unsigned short* __restrict__ Kt,
                                                  const unsigned short* __restrict__ Vc,
                                                  unsigned short* __restrict__ Ot) {
  __shared__ unsigned short sK[64 * 64];        // [key][dd]
  __shared__ unsigned short sV[64 * 64];        // [dd][key]
  __shared__ unsigned short sP[8 * 16 * 64];    // wave-private P stash [q][key]

  const int tid  = threadIdx.x;
  const int w    = tid >> 5;
  const int lane = tid & 31;
  const int li   = lane & 15;
  const int lh   = lane >> 4;
  const int h    = blockIdx.y;
  const int q0   = blockIdx.x * 128 + w * 16;

  bf16x16 qf[2];
#pragma unroll
  for (int kk = 0; kk < 2; ++kk)
    qf[kk] = load_frag(Qt + (size_t)(q0 + li) * EMBED + h * HDIM, kk * 32, lh);

  float m_r[8], l_r[8];
  f32x8 o_acc[4];
#pragma unroll
  for (int r = 0; r < 8; ++r) { m_r[r] = -1e30f; l_r[r] = 0.f; }
#pragma unroll
  for (int t = 0; t < 4; ++t) o_acc[t] = zero8();

  unsigned short* myP = sP + w * (16 * 64);

  for (int kb = 0; kb < NTOK; kb += 64) {
    __syncthreads();
    // cooperative load of 64-key K and V tiles; async DMA to LDS on gfx1250
    for (int c = tid; c < 512; c += 256) {
      int row = c >> 3, c8 = (c & 7) * 8;
      const unsigned short* gk = Kt + (size_t)(kb + row) * EMBED + h * HDIM + c8;
      const unsigned short* gv = Vc + (size_t)(h * HDIM + row) * NTOK + kb + c8;
      unsigned short* lk = sK + row * 64 + c8;
      unsigned short* lv = sV + row * 64 + c8;
#if HAVE_ASYNC_LDS
      __builtin_amdgcn_global_load_async_to_lds_b128((v4i_glb*)gk, (v4i_lds*)lk, 0, 0);
      __builtin_amdgcn_global_load_async_to_lds_b128((v4i_glb*)gv, (v4i_lds*)lv, 0, 0);
#else
      *reinterpret_cast<uint4*>(lk) = *reinterpret_cast<const uint4*>(gk);
      *reinterpret_cast<uint4*>(lv) = *reinterpret_cast<const uint4*>(gv);
#endif
      if (kb + 64 < NTOK) {                 // prefetch next tile into cache
        __builtin_prefetch(gk + (size_t)64 * EMBED, 0, 1);
        __builtin_prefetch(gv + 64, 0, 1);
      }
    }
#if HAVE_ASYNC_LDS
#if __has_builtin(__builtin_amdgcn_s_wait_asynccnt)
    __builtin_amdgcn_s_wait_asynccnt(0);
#else
    asm volatile("s_wait_asynccnt 0" ::: "memory");
#endif
#endif
    __syncthreads();

    // S = Q * K^T  (16 x 64, f32)
    f32x8 s[4];
#pragma unroll
    for (int nt = 0; nt < 4; ++nt) {
      f32x8 a = zero8();
#pragma unroll
      for (int kk = 0; kk < 2; ++kk) {
        bf16x16 kf = load_frag(sK + (nt * 16 + li) * 64, kk * 32, lh);
        a = wmma_bf16(qf[kk], kf, a);
      }
      s[nt] = a;
    }

    // online softmax: C-tile row r lives on one 16-lane half; reduce with shfl_xor
#pragma unroll
    for (int r = 0; r < 8; ++r) {
      float v = fmaxf(fmaxf(s[0][r], s[1][r]), fmaxf(s[2][r], s[3][r]));
#pragma unroll
      for (int off = 1; off < 16; off <<= 1) v = fmaxf(v, __shfl_xor(v, off, 16));
      float mnew  = fmaxf(m_r[r], v);
      float alpha = __expf(m_r[r] - mnew);
      m_r[r] = mnew;
      l_r[r] *= alpha;
#pragma unroll
      for (int t = 0; t < 4; ++t) o_acc[t][r] = o_acc[t][r] * alpha;
      float ps = 0.f;
#pragma unroll
      for (int nt = 0; nt < 4; ++nt) {
        float p = __expf(s[nt][r] - mnew);
        s[nt][r] = p;
        ps += p;
      }
#pragma unroll
      for (int off = 1; off < 16; off <<= 1) ps += __shfl_xor(ps, off, 16);
      l_r[r] += ps;
    }

    // stash P (C layout -> [q][key] in LDS); same-wave DS ops are in-order
#pragma unroll
    for (int nt = 0; nt < 4; ++nt)
#pragma unroll
      for (int r = 0; r < 8; ++r)
        myP[(lh * 8 + r) * 64 + nt * 16 + li] = f2bf(s[nt][r]);

    // O += P * V
    bf16x16 pa[2];
#pragma unroll
    for (int kk = 0; kk < 2; ++kk) pa[kk] = load_frag(myP + li * 64, kk * 32, lh);
#pragma unroll
    for (int dt = 0; dt < 4; ++dt)
#pragma unroll
      for (int kk = 0; kk < 2; ++kk) {
        bf16x16 vf = load_frag(sV + (dt * 16 + li) * 64, kk * 32, lh);
        o_acc[dt] = wmma_bf16(pa[kk], vf, o_acc[dt]);
      }
  }

  // normalize, write Ot[n][c] bf16 (feeds final GEMM as B operand)
#pragma unroll
  for (int r = 0; r < 8; ++r) {
    float inv = 1.0f / l_r[r];
    int q = q0 + lh * 8 + r;
#pragma unroll
    for (int dt = 0; dt < 4; ++dt)
      Ot[(size_t)q * EMBED + h * HDIM + dt * 16 + li] = f2bf(o_acc[dt][r] * inv);
  }
}

// ---- launcher ------------------------------------------------------------

extern "C" void kernel_launch(void* const* d_in, const int* in_sizes, int n_in,
                              void* d_out, int out_size, void* d_ws, size_t ws_size,
                              hipStream_t stream) {
  (void)in_sizes; (void)n_in; (void)out_size; (void)ws_size;
  const float* hidden = (const float*)d_in[0];
  const float* q_w = (const float*)d_in[1];
  const float* q_b = (const float*)d_in[2];
  const float* k_w = (const float*)d_in[3];
  const float* k_b = (const float*)d_in[4];
  const float* v_w = (const float*)d_in[5];
  const float* v_b = (const float*)d_in[6];
  const float* o_w = (const float*)d_in[7];
  const float* o_b = (const float*)d_in[8];

  char* ws = (char*)d_ws;
  size_t off = 0;
  auto alloc = [&](size_t bytes) -> void* {
    void* p = ws + off;
    off = (off + bytes + 255) & ~(size_t)255;
    return p;
  };
  unsigned short* Xt = (unsigned short*)alloc((size_t)NTOK * EMBED * 2);
  unsigned short* Wq = (unsigned short*)alloc((size_t)EMBED * EMBED * 2);
  unsigned short* Wk = (unsigned short*)alloc((size_t)EMBED * EMBED * 2);
  unsigned short* Wv = (unsigned short*)alloc((size_t)EMBED * EMBED * 2);
  unsigned short* Wo = (unsigned short*)alloc((size_t)EMBED * EMBED * 2);
  unsigned short* Qt = (unsigned short*)alloc((size_t)NTOK * EMBED * 2);
  unsigned short* Kt = (unsigned short*)alloc((size_t)NTOK * EMBED * 2);
  unsigned short* Vc = (unsigned short*)alloc((size_t)NTOK * EMBED * 2);
  unsigned short* Ot = (unsigned short*)alloc((size_t)NTOK * EMBED * 2);

  const float SCALE = 0.125f;  // HEAD_DIM^-0.5, folded into k projection

  transpose_convert_k<<<(NTOK * EMBED) / 256, 256, 0, stream>>>(hidden, Xt);
  convert_scale_k<<<(EMBED * EMBED) / 256, 256, 0, stream>>>(q_w, Wq, EMBED * EMBED, 1.0f);
  convert_scale_k<<<(EMBED * EMBED) / 256, 256, 0, stream>>>(k_w, Wk, EMBED * EMBED, SCALE);
  convert_scale_k<<<(EMBED * EMBED) / 256, 256, 0, stream>>>(v_w, Wv, EMBED * EMBED, 1.0f);
  convert_scale_k<<<(EMBED * EMBED) / 256, 256, 0, stream>>>(o_w, Wo, EMBED * EMBED, 1.0f);

  dim3 g(EMBED / 128, NTOK / 128);
  gemm_wmma<0><<<g, 256, 0, stream>>>(Wq, Xt, q_b, 1.0f, Qt);
  gemm_wmma<0><<<g, 256, 0, stream>>>(Wk, Xt, k_b, SCALE, Kt);
  gemm_wmma<1><<<g, 256, 0, stream>>>(Wv, Xt, v_b, 1.0f, Vc);

  flash_attn<<<dim3(NTOK / 128, HEADS), 256, 0, stream>>>(Qt, Kt, Vc, Ot);

  gemm_wmma<2><<<g, 256, 0, stream>>>(Wo, Ot, o_b, 1.0f, d_out);
}
